// WukongLayer_47691316854854
// MI455X (gfx1250) — matrix operand order repaired and checked by reading, compiled, not run on hardware
//
#include <hip/hip_runtime.h>

// ---------------------------------------------------------------------------
// MI455X (gfx1250) implementation.
// Roofline: ~28 GFLOP of GEMM vs ~230 MB of HBM traffic -> memory floor
// ~10us @ 23.3 TB/s. f16 WMMA (v_wmma_f32_16x16x32_f16, f32 accumulate) for
// the 3 MLP GEMMs keeps compute below the memory floor; everything feeding
// the final LayerNorm outside the MLP stays fp32.
// Workspace layout (~95.5 MB): outbuf(f32 concat of fmb|lcb), fmn/h1/h2 (f16),
// w1T/w2T/w3T (f16, transposed), diag (64 f32).
// ---------------------------------------------------------------------------

typedef __attribute__((ext_vector_type(16))) _Float16 v16h;
typedef __attribute__((ext_vector_type(8)))  float    v8f;

union Frag { v16h v; uint4 u[2]; };

__device__ __forceinline__ float clip1(float v) { return fminf(fmaxf(v, -1.f), 1.f); }

// ---------------------------------------------------------------------------
// Weight transpose + fp32 -> f16 convert:  wT[n*K + k] = (f16) w[k*N + n]
// ---------------------------------------------------------------------------
__global__ __launch_bounds__(256) void transcvt_kernel(
    const float* __restrict__ w, _Float16* __restrict__ wT, int K, int N)
{
    __shared__ float tile[32][33];
    const int tx = threadIdx.x & 31;
    const int ty = threadIdx.x >> 5;
    const int n0 = blockIdx.x * 32;
    const int k0 = blockIdx.y * 32;
    #pragma unroll
    for (int i = 0; i < 32; i += 8)
        tile[ty + i][tx] = w[(size_t)(k0 + ty + i) * N + n0 + tx];
    __syncthreads();
    #pragma unroll
    for (int i = 0; i < 32; i += 8)
        wT[(size_t)(n0 + ty + i) * K + k0 + tx] = (_Float16)tile[tx][ty + i];
}

// ---------------------------------------------------------------------------
// Stage 1 (one block per batch element):
//   r = x^T @ w_rank (128x24), fm = x @ r (64x24), clipped-std LN -> fmn (f16)
//   lcb = w_lcb^T @ x (32x128) -> outbuf rows 32..63 (fp32)
// ---------------------------------------------------------------------------
__global__ __launch_bounds__(256) void stage1_kernel(
    const float* __restrict__ x, const float* __restrict__ w_lcb,
    const float* __restrict__ w_rank, _Float16* __restrict__ fmn,
    float* __restrict__ outbuf)
{
    __shared__ float smem[14464];              // 57.9 KB
    float* xs  = smem;                         // [64][129] padded x tile
    float* wr  = smem + 8256;                  // [64][24]
    float* rr  = smem + 8256 + 1536;           // [128][24]; later reused for w_lcb
    float* fm  = rr + 3072;                    // [1536]
    float* red = fm + 1536;                    // [16]

    const int t = threadIdx.x;
    const int b = blockIdx.x;

    const float4* xb4 = (const float4*)(x + (size_t)b * 8192);
    for (int i = t; i < 2048; i += 256) {
        float4 v = xb4[i];
        int e = i * 4, n = e >> 7, d = e & 127;
        float* p = xs + n * 129 + d;
        p[0] = v.x; p[1] = v.y; p[2] = v.z; p[3] = v.w;
    }
    for (int i = t; i < 64 * 24; i += 256) wr[i] = w_rank[i];
    __syncthreads();

    // r[d][j] = sum_n x[n][d] * w_rank[n][j]
    for (int i = t; i < 128 * 24; i += 256) {
        int d = i / 24, j = i - d * 24;
        float s = 0.f;
        #pragma unroll 16
        for (int n = 0; n < 64; ++n) s = fmaf(xs[n * 129 + d], wr[n * 24 + j], s);
        rr[i] = s;
    }
    __syncthreads();

    // fm[n][j] = sum_d x[n][d] * r[d][j]
    for (int i = t; i < 1536; i += 256) {
        int n = i / 24, j = i - n * 24;
        float s = 0.f;
        #pragma unroll 16
        for (int d = 0; d < 128; ++d) s = fmaf(xs[n * 129 + d], rr[d * 24 + j], s);
        fm[i] = s;
    }
    __syncthreads();

    // LayerNorm over 1536 (no affine, std clipped to [1e-11, 1e7])
    float ls = 0.f, lq = 0.f;
    for (int i = t; i < 1536; i += 256) { float v = fm[i]; ls += v; lq = fmaf(v, v, lq); }
    for (int o = 16; o; o >>= 1) { ls += __shfl_xor(ls, o, 32); lq += __shfl_xor(lq, o, 32); }
    if ((t & 31) == 0) { red[t >> 5] = ls; red[8 + (t >> 5)] = lq; }
    __syncthreads();
    float S = 0.f, Q = 0.f;
    #pragma unroll
    for (int w = 0; w < 8; ++w) { S += red[w]; Q += red[8 + w]; }
    float mu  = S * (1.f / 1536.f);
    float var = fmaxf(Q * (1.f / 1536.f) - mu * mu, 0.f);
    float sd  = fminf(fmaxf(sqrtf(var), 1e-11f), 1e7f);
    float inv = 1.f / sd;
    _Float16* fout = fmn + (size_t)b * 1536;
    for (int i = t; i < 1536; i += 256) fout[i] = (_Float16)((fm[i] - mu) * inv);

    // reuse rr region for w_lcb (2048 floats <= 3072)
    float* wl = rr;
    for (int i = t; i < 64 * 32; i += 256) wl[i] = w_lcb[i];
    __syncthreads();

    // lcb[m][d] -> outbuf rows 32..63 of the concat
    float* ob = outbuf + (size_t)b * 8192 + 4096;
    for (int i = t; i < 32 * 128; i += 256) {
        int m = i >> 7, d = i & 127;
        float s = 0.f;
        #pragma unroll 16
        for (int n = 0; n < 64; ++n) s = fmaf(xs[n * 129 + d], wl[n * 32 + m], s);
        ob[i] = s;
    }
}

// ---------------------------------------------------------------------------
// f16 WMMA GEMM: C[M,N] = A[M,K] @ BT[N,K]^T, f32 accumulate.
// 128x128 block tile, 8 waves (2x4), wave tile 64x32 = 4x2 WMMA 16x16x32.
// Fragment loads follow ISA 7.12.2 16-bit layouts:
//   A: lane<16 -> row M=lane%16, K {0..7,16..23}; lane>=16 -> K {8..15,24..31}
//   B: lane<16 -> col N=lane%16, K {0..7,16..23}; lane>=16 -> K {8..15,24..31}
// ---------------------------------------------------------------------------
template<int RELU, int OUTF16>
__global__ __launch_bounds__(256) void gemm_f16_kernel(
    const _Float16* __restrict__ A, const _Float16* __restrict__ BT,
    float* __restrict__ Cf, _Float16* __restrict__ Ch,
    int M, int N, int K, int out_stride)
{
    __shared__ _Float16 sA[128 * 40];   // 128 rows x 32 k, pad 8 (80B stride)
    __shared__ _Float16 sB[128 * 40];
    const int t    = threadIdx.x;
    const int lane = t & 31;
    const int wave = t >> 5;
    const int wm   = wave >> 2;       // 0..1
    const int wn   = wave & 3;        // 0..3
    const int m0   = blockIdx.y * 128;
    const int n0   = blockIdx.x * 128;
    const int lr   = lane & 15;
    const int hi   = lane >> 4;       // 0 or 1
    (void)M; (void)N;

    v8f acc[4][2] = {};

    for (int kt = 0; kt < K; kt += 32) {
        // stage tiles: 512 x uint4 (8 f16) chunks each, 2 chunks per thread
        #pragma unroll
        for (int it = 0; it < 2; ++it) {
            int c = t + it * 256;
            int row = c >> 2, kc = (c & 3) << 3;
            *(uint4*)(sA + row * 40 + kc) = *(const uint4*)(A + (size_t)(m0 + row) * K + kt + kc);
            *(uint4*)(sB + row * 40 + kc) = *(const uint4*)(BT + (size_t)(n0 + row) * K + kt + kc);
        }
        __syncthreads();

        Frag a[4], bf[2];
        #pragma unroll
        for (int i = 0; i < 4; ++i) {
            const _Float16* p = sA + (wm * 64 + i * 16 + lr) * 40 + hi * 8;
            a[i].u[0] = *(const uint4*)(p);
            a[i].u[1] = *(const uint4*)(p + 16);
        }
        #pragma unroll
        for (int j = 0; j < 2; ++j) {
            const _Float16* p = sB + (wn * 32 + j * 16 + lr) * 40 + hi * 8;
            bf[j].u[0] = *(const uint4*)(p);
            bf[j].u[1] = *(const uint4*)(p + 16);
        }
        #pragma unroll
        for (int i = 0; i < 4; ++i)
            #pragma unroll
            for (int j = 0; j < 2; ++j)
                acc[i][j] = __builtin_amdgcn_wmma_f32_16x16x32_f16(
                    false, a[i].v, false, bf[j].v, (short)0, acc[i][j], false, false);
        __syncthreads();
    }

    // C layout: VGPR e -> M = e + (lane<16 ? 0 : 8), N = lane%16
    #pragma unroll
    for (int i = 0; i < 4; ++i) {
        #pragma unroll
        for (int j = 0; j < 2; ++j) {
            int col = n0 + wn * 32 + j * 16 + lr;
            #pragma unroll
            for (int e = 0; e < 8; ++e) {
                int row = m0 + wm * 64 + i * 16 + e + hi * 8;
                float v = acc[i][j][e];
                if (RELU) v = fmaxf(v, 0.f);
                size_t idx = (size_t)row * out_stride + col;
                if (OUTF16) Ch[idx] = (_Float16)v;
                else        Cf[idx] = v;
            }
        }
    }
}

// ---------------------------------------------------------------------------
// diag[n] = clip(sum_d clip(out[0,n,d])^2)  (depends only on batch 0)
// ---------------------------------------------------------------------------
__global__ void diag_kernel(const float* __restrict__ outbuf, float* __restrict__ diag)
{
    int n = threadIdx.x;
    if (n < 64) {
        const float* p = outbuf + n * 128;
        float s = 0.f;
        for (int d = 0; d < 128; ++d) { float v = clip1(p[d]); s = fmaf(v, v, s); }
        diag[n] = clip1(s);
    }
}

// ---------------------------------------------------------------------------
// Epilogue: one wave per (b,n) row of 128. 27-term "extra" (x1e-10), residual,
// final LayerNorm (eps=1e-3) with gamma/beta.
// ---------------------------------------------------------------------------
__global__ __launch_bounds__(256) void epilogue_kernel(
    const float* __restrict__ outbuf, const float* __restrict__ x,
    const float* __restrict__ noise, const float* __restrict__ gamma,
    const float* __restrict__ beta, const float* __restrict__ diag,
    float* __restrict__ out)
{
    const int lane = threadIdx.x & 31;
    const int row  = blockIdx.x * 8 + (threadIdx.x >> 5);   // b*64 + n
    const int b    = row >> 6;
    const int n    = row & 63;
    const int d0   = lane * 4;

    float4 o4 = *(const float4*)(outbuf + (size_t)b * 8192 + n * 128 + d0);
    float4 x4 = *(const float4*)(x     + (size_t)row * 128 + d0);
    float4 z4 = *(const float4*)(noise + (size_t)row * 128 + d0);
    float o[4]  = {o4.x, o4.y, o4.z, o4.w};
    float xv[4] = {x4.x, x4.y, x4.z, x4.w};
    float zv[4] = {z4.x, z4.y, z4.z, z4.w};

    float s[4], e[4];
    float sumE = 0.f, mx = -3.402823e38f, pr = 1.f, gs = 0.f;
    #pragma unroll
    for (int j = 0; j < 4; ++j) {
        s[j] = clip1(o[j]);
        e[j] = expf(s[j]);
        sumE += e[j];
        mx = fmaxf(mx, s[j]);
        pr *= s[j];
        gs += s[j];
    }
    // group-of-8 conv sums: lane pair (2l, 2l+1) covers one group of 8
    float g8 = gs + __shfl_xor(gs, 1, 32);
    for (int off = 16; off; off >>= 1) {
        sumE += __shfl_xor(sumE, off, 32);
        mx    = fmaxf(mx, __shfl_xor(mx, off, 32));
        pr   *= __shfl_xor(pr, off, 32);
    }
    float invSum = 1.f / sumE;
    float prC = clip1(pr), mxC = clip1(mx), g8C = clip1(g8), dn = diag[n];

    float pre[4];
    float ps = 0.f;
    #pragma unroll
    for (int j = 0; j < 4; ++j) {
        int d = d0 + j;
        float sv = s[j], av = fabsf(sv), ev = e[j];
        float sgn  = (sv > 0.f) ? 1.f : ((sv < 0.f) ? -1.f : 0.f);
        float pos1 = (sv >= 0.f) ? 1.f : 0.f;
        float extra =
              clip1(logf(av + 1.f))                    // log(a+1)
            + clip1(rsqrtf(av + 1.f))                  // rsqrt(a+1)
            + clip1(ev * invSum)                       // softmax
            + clip1(tanhf(sv))                         // tanh
            + clip1((sv >= 0.f) ? sv : 0.2f * sv)      // leaky relu
            + clip1(ev)                                // exp
            + clip1(log1pf(ev))                        // softplus
            + clip1(log1pf(av))                        // log1p
            + 0.f                                      // zeros_like
            + clip1(rintf(sv))                         // round (half-even)
            + 1.f                                      // ones_like
            + sgn                                      // sign
            + prC                                      // prod over row
            + ((sv != sv) ? 1.f : 0.f)                 // isnan
            + pos1                                     // ~(safe<0)
            + ((sv != 0.f) ? 1.f : 0.f)                // safe != 0
            + pos1                                     // safe >= 0
            + clip1(truncf(sv))                        // int8 cast roundtrip
            + mxC                                      // max over row
            + clip1(2.f * sv) + clip1(2.f * sv)        // add_n + Add
            + 7.f * sv                                 // identity stack
            + clip1((sv > 0.5f) ? -1.f : sv)           // scatter update
            + g8C                                      // 1x1 group conv
            + clip1(zv[j])                             // noise
            + ((d == 0) ? 0.f : 1.f)                   // clip(arange(D))
            + dn;                                      // diag(matmul(safe0,safe0^T))
        pre[j] = o[j] + xv[j] + 1e-10f * extra;
        ps += pre[j];
    }

    for (int off = 16; off; off >>= 1) ps += __shfl_xor(ps, off, 32);
    float mean = ps * (1.f / 128.f);
    float qs = 0.f;
    #pragma unroll
    for (int j = 0; j < 4; ++j) { float c = pre[j] - mean; qs = fmaf(c, c, qs); }
    for (int off = 16; off; off >>= 1) qs += __shfl_xor(qs, off, 32);
    float inv = rsqrtf(qs * (1.f / 128.f) + 1e-3f);

    float4 r;
    r.x = (pre[0] - mean) * inv * gamma[d0]     + beta[d0];
    r.y = (pre[1] - mean) * inv * gamma[d0 + 1] + beta[d0 + 1];
    r.z = (pre[2] - mean) * inv * gamma[d0 + 2] + beta[d0 + 2];
    r.w = (pre[3] - mean) * inv * gamma[d0 + 3] + beta[d0 + 3];
    *(float4*)(out + (size_t)row * 128 + d0) = r;
}

// ---------------------------------------------------------------------------
extern "C" void kernel_launch(void* const* d_in, const int* in_sizes, int n_in,
                              void* d_out, int out_size, void* d_ws, size_t ws_size,
                              hipStream_t stream)
{
    (void)in_sizes; (void)n_in; (void)out_size; (void)ws_size;
    const float* x      = (const float*)d_in[0];
    const float* noise  = (const float*)d_in[1];
    const float* w_lcb  = (const float*)d_in[2];
    const float* w_rank = (const float*)d_in[3];
    const float* w1     = (const float*)d_in[4];
    const float* w2     = (const float*)d_in[5];
    const float* w3     = (const float*)d_in[6];
    const float* gamma  = (const float*)d_in[7];
    const float* beta   = (const float*)d_in[8];
    float* out = (float*)d_out;

    char* ws = (char*)d_ws;
    size_t off = 0;
    auto carve = [&](size_t bytes) -> void* {
        void* p = ws + off;
        off += (bytes + 255) & ~(size_t)255;
        return p;
    };
    float*    outbuf = (float*)   carve((size_t)2048 * 8192 * 4);  // 64 MB concat fmb|lcb
    _Float16* fmn    = (_Float16*)carve((size_t)2048 * 1536 * 2);
    _Float16* h1     = (_Float16*)carve((size_t)2048 * 1024 * 2);
    _Float16* h2     = (_Float16*)carve((size_t)2048 * 1024 * 2);
    _Float16* w1T    = (_Float16*)carve((size_t)1536 * 1024 * 2);
    _Float16* w2T    = (_Float16*)carve((size_t)1024 * 1024 * 2);
    _Float16* w3T    = (_Float16*)carve((size_t)1024 * 4096 * 2);
    float*    diag   = (float*)   carve(64 * 4);

    // 1) convert + transpose weights (N-major f16 for contiguous B fragments)
    transcvt_kernel<<<dim3(1024 / 32, 1536 / 32), 256, 0, stream>>>(w1, w1T, 1536, 1024);
    transcvt_kernel<<<dim3(1024 / 32, 1024 / 32), 256, 0, stream>>>(w2, w2T, 1024, 1024);
    transcvt_kernel<<<dim3(4096 / 32, 1024 / 32), 256, 0, stream>>>(w3, w3T, 1024, 4096);
    // 2) per-batch factorization + LN + lcb
    stage1_kernel<<<2048, 256, 0, stream>>>(x, w_lcb, w_rank, fmn, outbuf);
    // 3) MLP via f16 WMMA
    gemm_f16_kernel<1, 1><<<dim3(1024 / 128, 2048 / 128), 256, 0, stream>>>(
        fmn, w1T, nullptr, h1, 2048, 1024, 1536, 1024);
    gemm_f16_kernel<1, 1><<<dim3(1024 / 128, 2048 / 128), 256, 0, stream>>>(
        h1, w2T, nullptr, h2, 2048, 1024, 1024, 1024);
    gemm_f16_kernel<0, 0><<<dim3(4096 / 128, 2048 / 128), 256, 0, stream>>>(
        h2, w3T, outbuf, nullptr, 2048, 4096, 1024, 8192);   // rows 0..31 of concat
    // 4) batch-0 diag term, 5) epilogue + final LN
    diag_kernel<<<1, 64, 0, stream>>>(outbuf, diag);
    epilogue_kernel<<<2048 * 64 / 8, 256, 0, stream>>>(outbuf, x, noise, gamma, beta, diag, out);
}
